// MambaBlock_47785806135916
// MI455X (gfx1250) — matrix operand order repaired and checked by reading, not compiled, not run on hardware
//
#include <hip/hip_runtime.h>
#include <hip/hip_bf16.h>
#include <math.h>

// ---------------------------------------------------------------------------
// Mamba block forward for gfx1250 (MI455X).
// Large GEMMs: bf16 WMMA, 64x128 block tile, LDS double-buffered staging via
// async global->LDS copies (ASYNCcnt), 32x32 register tile per wave.
// ---------------------------------------------------------------------------

#define D_MODEL 1024
#define D_INNER 1024
#define BATCH   2
#define SEQ     1024
#define NSTATE  16
#define ROWS    (BATCH * SEQ)      // 2048
#define ZCOLS   (2 * D_INNER)      // 2048

typedef __attribute__((ext_vector_type(16))) __bf16 bf16x16;
typedef __attribute__((ext_vector_type(8)))  __bf16 bf16x8;
typedef __attribute__((ext_vector_type(8)))  float  f32x8;
typedef __attribute__((ext_vector_type(4)))  float  f32x4;

// GCC-style vector matching the builtin prototype (V4i)
typedef int i32x4 __attribute__((vector_size(16)));
typedef __attribute__((address_space(1))) i32x4 as1_i32x4;
typedef __attribute__((address_space(3))) i32x4 as3_i32x4;

// Epilogue modes
#define EPI_F32          0   // Cf[row,col] = acc
#define EPI_BIAS_BF16    1   // Cb[row,col] = bf16(acc + bias[col])
#define EPI_SOFTPLUS_F32 2   // Cf[row,col] = softplus(acc)
#define EPI_RESID_F32    3   // Cf[row,col] = acc + resid[row,col]

// ---------------- async global->LDS copy (gfx1250 ASYNCcnt path) -----------
#if defined(__gfx1250__) && __has_builtin(__builtin_amdgcn_global_load_async_to_lds_b128)
#define HAVE_ASYNC_LDS 1
#else
#define HAVE_ASYNC_LDS 0
#endif

__device__ __forceinline__ void cp16_to_lds(const __bf16* gsrc, __bf16* ldst)
{
#if HAVE_ASYNC_LDS
    // Generic VA == as(1) VA for global pointers; generic LDS VA low 32 bits
    // are the LDS offset (ISA 10.2).
    as1_i32x4* g = (as1_i32x4*)(unsigned long long)(__SIZE_TYPE__)gsrc;
    as3_i32x4* l = (as3_i32x4*)(unsigned int)(__SIZE_TYPE__)ldst;
    __builtin_amdgcn_global_load_async_to_lds_b128(g, l, 0, 0);
#else
    *(f32x4*)ldst = *(const f32x4*)gsrc;   // sync fallback: b128 load + ds_store
#endif
}

template <int N>
__device__ __forceinline__ void wait_async_copies()
{
#if HAVE_ASYNC_LDS
#if __has_builtin(__builtin_amdgcn_s_wait_asynccnt)
    __builtin_amdgcn_s_wait_asynccnt(N);
#else
    asm volatile("s_wait_asynccnt %0" :: "n"(N));
#endif
#endif
}

// ---------------------------------------------------------------------------
// Big-N WMMA GEMM:  C[M,N] = A[M,K](bf16) * Bt[N,K](bf16)^T
// Block: 256 threads = 8 waves as 2(M) x 4(N); wave tile 32x32 (2x2 WMMA).
// Block tile 64(M) x 128(N); K stepped by 32, LDS double buffered.
// ---------------------------------------------------------------------------
#define BM 64
#define BN 128
#define BK 32
#define LPAD 8                    // +16B per row: spreads LDS banks
#define LSTR (BK + LPAD)          // 40 bf16 = 80B row stride (16B aligned)

__device__ __forceinline__ void stage_copy(
    const __bf16* __restrict__ A, const __bf16* __restrict__ Bt,
    int K, int blockM, int blockN, int k0,
    __bf16* Ash, __bf16* Bsh, int t)
{
    {   // A tile: 64 rows x 4 chunks of 16B = 256 chunks (1 per thread)
        int row = t >> 2, ch = t & 3;
        cp16_to_lds(A + (size_t)(blockM + row) * K + k0 + ch * 8,
                    Ash + row * LSTR + ch * 8);
    }
#pragma unroll
    for (int i = 0; i < 2; ++i) {  // B tile: 128 rows x 4 chunks = 512 chunks
        int u = t + i * 256;
        int row = u >> 2, ch = u & 3;
        cp16_to_lds(Bt + (size_t)(blockN + row) * K + k0 + ch * 8,
                    Bsh + row * LSTR + ch * 8);
    }
}

template <int EPI>
__global__ __launch_bounds__(256) void gemm_wmma_big(
    const __bf16* __restrict__ A, const __bf16* __restrict__ Bt,
    int M, int N, int K,
    float* __restrict__ Cf, __bf16* __restrict__ Cb,
    const float* __restrict__ bias, const float* __restrict__ resid)
{
    __shared__ __bf16 Ash[2][BM * LSTR];   // 2 x 5120 B
    __shared__ __bf16 Bsh[2][BN * LSTR];   // 2 x 10240 B

    const int t    = threadIdx.x;
    const int lane = t & 31;
    const int wave = t >> 5;
    const int wm   = wave >> 2;            // 0..1
    const int wn   = wave & 3;             // 0..3
    const int half = lane >> 4;            // 0/1
    const int lidx = lane & 15;
    const int blockM = blockIdx.x * BM;
    const int blockN = blockIdx.y * BN;

    f32x8 acc[2][2];
#pragma unroll
    for (int mt = 0; mt < 2; ++mt)
#pragma unroll
        for (int nt = 0; nt < 2; ++nt)
#pragma unroll
            for (int r = 0; r < 8; ++r)
                acc[mt][nt][r] = 0.f;

    const int KT = K / BK;
    stage_copy(A, Bt, K, blockM, blockN, 0, Ash[0], Bsh[0], t);

    for (int ks = 0; ks < KT; ++ks) {
        const int cur = ks & 1;
        if (ks + 1 < KT) {
            stage_copy(A, Bt, K, blockM, blockN, (ks + 1) * BK,
                       Ash[cur ^ 1], Bsh[cur ^ 1], t);
            // 3 async copies per thread per stage; in-order retirement =>
            // <=3 outstanding means the current stage is fully in LDS.
            wait_async_copies<3>();
        } else {
            wait_async_copies<0>();
        }
        __syncthreads();

        const __bf16* As = Ash[cur];
        const __bf16* Bs = Bsh[cur];

        bf16x16 av[2], bv[2];
#pragma unroll
        for (int mt = 0; mt < 2; ++mt) {
            const int row = (wm * 32 + mt * 16 + lidx) * LSTR;
            bf16x8 lo = *(const bf16x8*)(As + row + half * 8);
            bf16x8 hi = *(const bf16x8*)(As + row + 16 + half * 8);
            av[mt] = __builtin_shufflevector(lo, hi,
                0, 1, 2, 3, 4, 5, 6, 7, 8, 9, 10, 11, 12, 13, 14, 15);
        }
#pragma unroll
        for (int nt = 0; nt < 2; ++nt) {
            const int row = (wn * 32 + nt * 16 + lidx) * LSTR;
            bf16x8 lo = *(const bf16x8*)(Bs + row + half * 8);
            bf16x8 hi = *(const bf16x8*)(Bs + row + 16 + half * 8);
            bv[nt] = __builtin_shufflevector(lo, hi,
                0, 1, 2, 3, 4, 5, 6, 7, 8, 9, 10, 11, 12, 13, 14, 15);
        }
#pragma unroll
        for (int mt = 0; mt < 2; ++mt)
#pragma unroll
            for (int nt = 0; nt < 2; ++nt)
                acc[mt][nt] = __builtin_amdgcn_wmma_f32_16x16x32_bf16(
                    false, av[mt], false, bv[nt], (short)0, acc[mt][nt],
                    false, false);
        __syncthreads();
    }

#pragma unroll
    for (int mt = 0; mt < 2; ++mt)
#pragma unroll
        for (int nt = 0; nt < 2; ++nt) {
            const int row0 = blockM + wm * 32 + mt * 16 + half * 8;
            const int col  = blockN + wn * 32 + nt * 16 + lidx;
#pragma unroll
            for (int r = 0; r < 8; ++r) {
                const size_t idx = (size_t)(row0 + r) * N + col;
                float v = acc[mt][nt][r];
                if (EPI == EPI_F32) {
                    Cf[idx] = v;
                } else if (EPI == EPI_BIAS_BF16) {
                    Cb[idx] = (__bf16)(v + bias[col]);
                } else if (EPI == EPI_SOFTPLUS_F32) {
                    Cf[idx] = (v > 20.f) ? v : log1pf(expf(v));
                } else { // EPI_RESID_F32
                    Cf[idx] = v + resid[idx];
                }
            }
        }
}

// ---------------------------------------------------------------------------
// Narrow-N WMMA GEMM (N=32): 8 waves as 4(M) x 2(N), one 16x16 tile per wave,
// direct global loads (no reuse to exploit at one tile column).
// ---------------------------------------------------------------------------
__global__ __launch_bounds__(256) void gemm_wmma_narrow(
    const __bf16* __restrict__ A, const __bf16* __restrict__ Bt,
    int M, int N, int K, float* __restrict__ Cf)
{
    const int lane  = threadIdx.x & 31;
    const int wave  = threadIdx.x >> 5;
    const int tileM = blockIdx.x * 64 + (wave >> 1) * 16;
    const int tileN = (wave & 1) * 16;
    const int half  = lane >> 4;
    const int lidx  = lane & 15;

    const __bf16* __restrict__ arow = A  + (size_t)(tileM + lidx) * K;
    const __bf16* __restrict__ brow = Bt + (size_t)(tileN + lidx) * K;

    f32x8 acc = {0.f, 0.f, 0.f, 0.f, 0.f, 0.f, 0.f, 0.f};
    for (int k = 0; k < K; k += 32) {
        bf16x8 a0 = *(const bf16x8*)(arow + k + half * 8);
        bf16x8 a1 = *(const bf16x8*)(arow + k + 16 + half * 8);
        bf16x8 b0 = *(const bf16x8*)(brow + k + half * 8);
        bf16x8 b1 = *(const bf16x8*)(brow + k + 16 + half * 8);
        bf16x16 av = __builtin_shufflevector(a0, a1,
            0, 1, 2, 3, 4, 5, 6, 7, 8, 9, 10, 11, 12, 13, 14, 15);
        bf16x16 bv = __builtin_shufflevector(b0, b1,
            0, 1, 2, 3, 4, 5, 6, 7, 8, 9, 10, 11, 12, 13, 14, 15);
        acc = __builtin_amdgcn_wmma_f32_16x16x32_bf16(
            false, av, false, bv, (short)0, acc, false, false);
    }
    const int row0 = tileM + half * 8;
    const int col  = tileN + lidx;
#pragma unroll
    for (int r = 0; r < 8; ++r)
        Cf[(size_t)(row0 + r) * N + col] = acc[r];
}

// ---------------------------------------------------------------------------
// Transpose + f32->bf16 convert: src[R][C] f32 -> dst[C][R] bf16
// ---------------------------------------------------------------------------
__global__ __launch_bounds__(256) void transpose_cvt_kernel(
    const float* __restrict__ src, __bf16* __restrict__ dst, int R, int C)
{
    __shared__ float tile[32][33];
    const int c0 = blockIdx.x * 32, r0 = blockIdx.y * 32;
#pragma unroll
    for (int i = 0; i < 32; i += 8) {
        int r = r0 + threadIdx.y + i, c = c0 + threadIdx.x;
        tile[threadIdx.y + i][threadIdx.x] =
            (r < R && c < C) ? src[(size_t)r * C + c] : 0.f;
    }
    __syncthreads();
#pragma unroll
    for (int i = 0; i < 32; i += 8) {
        int c = c0 + threadIdx.y + i, r = r0 + threadIdx.x;
        if (c < C && r < R)
            dst[(size_t)c * R + r] = (__bf16)tile[threadIdx.x][threadIdx.y + i];
    }
}

__global__ __launch_bounds__(256) void cvt_f32_bf16_kernel(
    const float* __restrict__ src, __bf16* __restrict__ dst, int n)
{
    int i = blockIdx.x * blockDim.x + threadIdx.x;
    if (i < n) dst[i] = (__bf16)src[i];
}

__global__ __launch_bounds__(256) void pack_wbc_kernel(
    const float* __restrict__ WB, const float* __restrict__ WC,
    __bf16* __restrict__ dst)
{
    int i = blockIdx.x * blockDim.x + threadIdx.x;
    if (i >= 32 * D_INNER) return;
    int n = i / D_INNER, k = i % D_INNER;
    float v = (n < NSTATE) ? WB[k * NSTATE + n] : WC[k * NSTATE + (n - NSTATE)];
    dst[(size_t)n * D_INNER + k] = (__bf16)v;
}

// ---------------------------------------------------------------------------
// Depthwise conv1d (K=3, same padding) on the val half of z, then SiLU.
// ---------------------------------------------------------------------------
__global__ __launch_bounds__(256) void conv_silu_kernel(
    const float* __restrict__ z, const float* __restrict__ conv_w,
    float* __restrict__ valf, __bf16* __restrict__ valb)
{
    int idx = blockIdx.x * blockDim.x + threadIdx.x;
    if (idx >= ROWS * D_INNER) return;
    int d = idx % D_INNER;
    int l = (idx / D_INNER) % SEQ;
    size_t row = (size_t)(idx / D_INNER);

    float w0 = conv_w[d * 3 + 0];
    float w1 = conv_w[d * 3 + 1];
    float w2 = conv_w[d * 3 + 2];

    float s = w1 * z[row * ZCOLS + d];
    if (l > 0)       s += w0 * z[(row - 1) * ZCOLS + d];
    if (l < SEQ - 1) s += w2 * z[(row + 1) * ZCOLS + d];

    float sv = s / (1.f + expf(-s));
    valf[row * D_INNER + d] = sv;
    valb[row * D_INNER + d] = (__bf16)sv;
}

// ---------------------------------------------------------------------------
// Selective scan: one thread per (b,d) channel, 16 states in registers,
// sequential over L; fuses output gating and emits bf16.
// ---------------------------------------------------------------------------
__global__ __launch_bounds__(256) void scan_kernel(
    const float* __restrict__ Delta, const float* __restrict__ BC,
    const float* __restrict__ val, const float* __restrict__ z,
    const float* __restrict__ A_log, __bf16* __restrict__ gy)
{
    int idx = blockIdx.x * blockDim.x + threadIdx.x;
    if (idx >= BATCH * D_INNER) return;
    int b = idx / D_INNER, d = idx % D_INNER;

    float a[NSTATE], h[NSTATE];
#pragma unroll
    for (int n = 0; n < NSTATE; ++n) {
        a[n] = -expf(A_log[d * NSTATE + n]);
        h[n] = 0.f;
    }

    for (int l = 0; l < SEQ; ++l) {
        size_t row   = (size_t)b * SEQ + l;
        float  delta = Delta[row * D_MODEL + d];
        float  u     = val[row * D_INNER + d];
        const float* __restrict__ bc = BC + row * 32;
        float yv = 0.f;
#pragma unroll
        for (int n = 0; n < NSTATE; ++n) {
            float dA = delta * a[n];
            float Ab = expf(dA);
            float Bb = expm1f(dA) / (dA + 1e-6f) * delta * bc[n];
            h[n] = Ab * h[n] + Bb * u;
            yv  += h[n] * bc[NSTATE + n];
        }
        float g  = z[row * ZCOLS + D_INNER + d];
        float sg = g / (1.f + expf(-g));
        gy[row * D_INNER + d] = (__bf16)(yv * sg);
    }
}

// ---------------------------------------------------------------------------
extern "C" void kernel_launch(void* const* d_in, const int* in_sizes, int n_in,
                              void* d_out, int out_size, void* d_ws, size_t ws_size,
                              hipStream_t stream)
{
    (void)in_sizes; (void)n_in; (void)out_size; (void)ws_size;

    const float* x          = (const float*)d_in[0];
    const float* W_in       = (const float*)d_in[1];
    const float* conv_w     = (const float*)d_in[2];
    const float* W_B        = (const float*)d_in[3];
    const float* W_C        = (const float*)d_in[4];
    const float* W_delta    = (const float*)d_in[5];
    const float* delta_base = (const float*)d_in[6];
    const float* W_tau      = (const float*)d_in[7];
    const float* A_log      = (const float*)d_in[8];
    const float* W_out      = (const float*)d_in[9];
    float* out = (float*)d_out;

    char*  ws  = (char*)d_ws;
    size_t off = 0;
    auto alloc = [&](size_t bytes) -> char* {
        char* p = ws + off;
        off = (off + bytes + 255) & ~(size_t)255;
        return p;
    };

    float*  z      = (float*)alloc((size_t)ROWS * ZCOLS * 4);
    float*  valf   = (float*)alloc((size_t)ROWS * D_INNER * 4);
    float*  DeltaB = (float*)alloc((size_t)ROWS * D_MODEL * 4);
    float*  BC     = (float*)alloc((size_t)ROWS * 32 * 4);
    __bf16* xb     = (__bf16*)alloc((size_t)ROWS * D_MODEL * 2);
    __bf16* valb   = (__bf16*)alloc((size_t)ROWS * D_INNER * 2);
    __bf16* cdb    = (__bf16*)alloc((size_t)ROWS * D_MODEL * 2);
    __bf16* gyb    = (__bf16*)alloc((size_t)ROWS * D_INNER * 2);
    __bf16* Wint   = (__bf16*)alloc((size_t)ZCOLS * D_MODEL * 2);
    __bf16* Wbct   = (__bf16*)alloc((size_t)32 * D_INNER * 2);
    __bf16* Wdt    = (__bf16*)alloc((size_t)D_MODEL * D_INNER * 2);
    __bf16* Wtt    = (__bf16*)alloc((size_t)D_MODEL * D_MODEL * 2);
    __bf16* Wot    = (__bf16*)alloc((size_t)D_MODEL * D_INNER * 2);

    const dim3 tblk(32, 8);

    // weight preprocessing: transpose + convert to bf16 [N][K]
    transpose_cvt_kernel<<<dim3(ZCOLS / 32, D_MODEL / 32), tblk, 0, stream>>>(
        W_in, Wint, D_MODEL, ZCOLS);
    transpose_cvt_kernel<<<dim3(D_MODEL / 32, D_INNER / 32), tblk, 0, stream>>>(
        W_delta, Wdt, D_INNER, D_MODEL);
    transpose_cvt_kernel<<<dim3(D_MODEL / 32, D_MODEL / 32), tblk, 0, stream>>>(
        W_tau, Wtt, D_MODEL, D_MODEL);
    transpose_cvt_kernel<<<dim3(D_MODEL / 32, D_INNER / 32), tblk, 0, stream>>>(
        W_out, Wot, D_INNER, D_MODEL);
    pack_wbc_kernel<<<(32 * D_INNER + 255) / 256, 256, 0, stream>>>(W_B, W_C, Wbct);

    cvt_f32_bf16_kernel<<<(ROWS * D_MODEL + 255) / 256, 256, 0, stream>>>(
        x, xb, ROWS * D_MODEL);

    // GEMM 1: z = x @ W_in  [2048 x 2048]
    gemm_wmma_big<EPI_F32><<<dim3(ROWS / BM, ZCOLS / BN), 256, 0, stream>>>(
        xb, Wint, ROWS, ZCOLS, D_MODEL, z, nullptr, nullptr, nullptr);

    // depthwise conv + SiLU
    conv_silu_kernel<<<(ROWS * D_INNER + 255) / 256, 256, 0, stream>>>(
        z, conv_w, valf, valb);

    // GEMM 2: [B_proj | C_proj] = val @ [W_B | W_C]  [2048 x 32]
    gemm_wmma_narrow<<<dim3(ROWS / 64, 1), 256, 0, stream>>>(
        valb, Wbct, ROWS, 32, D_INNER, BC);

    // GEMM 3: cdelta = val @ W_delta + delta_base -> bf16
    gemm_wmma_big<EPI_BIAS_BF16><<<dim3(ROWS / BM, D_MODEL / BN), 256, 0, stream>>>(
        valb, Wdt, ROWS, D_MODEL, D_INNER, nullptr, cdb, delta_base, nullptr);

    // GEMM 4: Delta = softplus(cdelta @ W_tau)
    gemm_wmma_big<EPI_SOFTPLUS_F32><<<dim3(ROWS / BM, D_MODEL / BN), 256, 0, stream>>>(
        cdb, Wtt, ROWS, D_MODEL, D_MODEL, DeltaB, nullptr, nullptr, nullptr);

    // selective scan (fused output gating)
    scan_kernel<<<(BATCH * D_INNER + 255) / 256, 256, 0, stream>>>(
        DeltaB, BC, valf, z, A_log, gyb);

    // GEMM 5: out = gy @ W_out + x
    gemm_wmma_big<EPI_RESID_F32><<<dim3(ROWS / BM, D_MODEL / BN), 256, 0, stream>>>(
        gyb, Wot, ROWS, D_MODEL, D_INNER, out, nullptr, nullptr, x);
}
